// GraphEncoder_80539226734597
// MI455X (gfx1250) — compile-verified, matrix operand order checked by reading, not run on hardware
//
#include <hip/hip_runtime.h>

#define NN 50000
#define NE 800000
#define INC 128
#define HIDC 64
#define OUTC 32
#define CDIV(a,b) (((a)+(b)-1)/(b))

typedef __attribute__((ext_vector_type(16))) _Float16 v16h;
typedef __attribute__((ext_vector_type(8)))  float    v8f;

// K index inside a 32-wide chunk for (half-slot h, lane) per CDNA5 16-bit
// A-matrix VGPR layout (mirrored for B with N in place of M).
__device__ __forceinline__ int kmap(int h, int lane) {
  return ((h & 8) ? 16 : 0) + ((lane & 16) ? 8 : 0) + (((h >> 1) & 3) << 1) + (h & 1);
}

// ---------------- utility fills ----------------
__global__ void k_fill_f32(float* __restrict__ p, float v, int n) {
  int i = blockIdx.x * blockDim.x + threadIdx.x;
  if (i < n) p[i] = v;
}
__global__ void k_fill_u32(unsigned* __restrict__ p, unsigned v, int n) {
  int i = blockIdx.x * blockDim.x + threadIdx.x;
  if (i < n) p[i] = v;
}
__global__ void k_bias_init(float* __restrict__ out, const float* __restrict__ b,
                            int rows, int C) {
  int i = blockIdx.x * blockDim.x + threadIdx.x;
  if (i < rows * C) out[i] = b[i % C];
}

// ---------------- WMMA fragment packing ----------------
// A[M,K] f32 row-major -> f16 fragments: P[((mt*KC+kc)*32+lane)*16+h]
__global__ void k_pack_a(const float* __restrict__ A, _Float16* __restrict__ P,
                         int M, int K) {
  int t = blockIdx.x * blockDim.x + threadIdx.x;
  if (t >= M * K) return;
  int h = t & 15;
  int lane = (t >> 4) & 31;
  int frag = t >> 9;
  int KC = K >> 5;
  int kc = frag % KC;
  int mt = frag / KC;
  int row = mt * 16 + (lane & 15);
  int k = kc * 32 + kmap(h, lane);
  P[t] = (_Float16)A[row * K + k];
}
// B[K,N] f32 row-major -> f16 fragments: P[((nt*KC+kc)*32+lane)*16+h]
__global__ void k_pack_b(const float* __restrict__ B, _Float16* __restrict__ P,
                         int N, int K) {
  int t = blockIdx.x * blockDim.x + threadIdx.x;
  if (t >= N * K) return;
  int h = t & 15;
  int lane = (t >> 4) & 31;
  int frag = t >> 9;
  int KC = K >> 5;
  int kc = frag % KC;
  int nt = frag / KC;
  int col = nt * 16 + (lane & 15);
  int k = kc * 32 + kmap(h, lane);
  P[t] = (_Float16)B[k * N + col];
}

// ---------------- packed WMMA GEMM ----------------
// D[M,N] = A@B from pre-packed f16 fragments. One wave per 16x16 tile,
// K-loop fully unrolled (K compile-time): b128 loads feeding v_wmma chain.
template <int K>
__global__ void k_wmma_gemm_packed(const v16h* __restrict__ PA,
                                   const v16h* __restrict__ PB,
                                   float* __restrict__ D, int M, int N) {
  constexpr int KC = K / 32;
  int wave = (blockIdx.x * blockDim.x + threadIdx.x) >> 5;
  int lane = threadIdx.x & 31;
  int ntiles = N >> 4;
  int mt = wave / ntiles;
  int nt = wave - mt * ntiles;
  if (mt * 16 >= M) return;                // uniform per wave -> EXEC all-1s
  const v16h* pa = PA + (size_t)mt * KC * 32 + lane;
  const v16h* pb = PB + (size_t)nt * KC * 32 + lane;
  v8f c = {};
#pragma unroll
  for (int kc = 0; kc < KC; ++kc) {
    v16h a = pa[kc * 32];
    v16h b = pb[kc * 32];
    c = __builtin_amdgcn_wmma_f32_16x16x32_f16(false, a, false, b, (short)0, c,
                                               false, false);
  }
#pragma unroll
  for (int r = 0; r < 8; ++r) {
    int orow = mt * 16 + r + ((lane & 16) ? 8 : 0);
    D[orow * N + nt * 16 + (lane & 15)] = c[r];
  }
}

// ---------------- GCN ----------------
__global__ void k_deg_scatter(float* __restrict__ deg, const int* __restrict__ dst,
                              const float* __restrict__ w, int E) {
  int e = blockIdx.x * blockDim.x + threadIdx.x;
  if (e < E) atomicAdd(&deg[dst[e]], w[e]);
}
__global__ void k_dinv(float* __restrict__ deg, int n) {
  int i = blockIdx.x * blockDim.x + threadIdx.x;
  if (i < n) { float d = deg[i]; deg[i] = d > 0.f ? rsqrtf(d) : 0.f; }
}
__global__ void k_edge_norm(float* __restrict__ nrm, const float* __restrict__ dinv,
                            const int* __restrict__ src, const int* __restrict__ dst,
                            const float* __restrict__ w, int E) {
  int e = blockIdx.x * blockDim.x + threadIdx.x;
  if (e < E) nrm[e] = dinv[src[e]] * w[e] * dinv[dst[e]];
}
__global__ void k_gcn_scatter(float* __restrict__ acc, const float* __restrict__ H0,
                              const float* __restrict__ nrm,
                              const int* __restrict__ src, const int* __restrict__ dst,
                              int E) {
  int t = blockIdx.x * blockDim.x + threadIdx.x;
  if (t < E * HIDC) {
    int e = t >> 6, c = t & 63;
    atomicAdd(&acc[dst[e] * HIDC + c], H0[src[e] * HIDC + c] * nrm[e]);
  }
}
__global__ void k_gcn_finish(float* __restrict__ H, const float* __restrict__ acc,
                             const float* __restrict__ H0, const float* __restrict__ dinv,
                             const float* __restrict__ bg) {
  int t = blockIdx.x * blockDim.x + threadIdx.x;
  if (t < NN * HIDC) {
    int i = t >> 6, c = t & 63;
    float di = dinv[i];
    float v = acc[t] + H0[t] * di * di + bg[c];   // self loop: norm = dinv*1*dinv
    H[t] = fmaxf(v, 0.f);
  }
}

// ---------------- GAT ----------------
__device__ __forceinline__ float lrelu(float v) { return v > 0.f ? v : 0.2f * v; }
__device__ __forceinline__ unsigned enc_f(float f) {
  unsigned u = __float_as_uint(f);
  return (u & 0x80000000u) ? ~u : (u | 0x80000000u);
}
__device__ __forceinline__ float dec_f(unsigned u) {
  return (u & 0x80000000u) ? __uint_as_float(u ^ 0x80000000u) : __uint_as_float(~u);
}

__global__ void k_scores(const float* __restrict__ Hm, const float* __restrict__ Hl,
                         const float* __restrict__ am_s, const float* __restrict__ am_d,
                         const float* __restrict__ al_s, const float* __restrict__ al_d,
                         float* __restrict__ ssm, float* __restrict__ sdm,
                         float* __restrict__ ssl, float* __restrict__ sdl) {
  int i = blockIdx.x * blockDim.x + threadIdx.x;
  if (i >= NN) return;
  const float* hm = Hm + i * OUTC;
  const float* hl = Hl + i * OUTC;
  float a = 0.f, b = 0.f, c = 0.f, d = 0.f;
#pragma unroll
  for (int k = 0; k < OUTC; ++k) {
    float m = hm[k], l = hl[k];
    a += m * am_s[k]; b += m * am_d[k];
    c += l * al_s[k]; d += l * al_d[k];
  }
  ssm[i] = a; sdm[i] = b; ssl[i] = c; sdl[i] = d;
}

__device__ __forceinline__ void edge_sd(int e, const int* src, const int* dst,
                                        int& s, int& d) {
  if (e < NE) { s = src[e]; d = dst[e]; } else { s = d = e - NE; }
}

__global__ void k_gat_max(const int* __restrict__ src, const int* __restrict__ dst,
                          const float* __restrict__ ssm, const float* __restrict__ sdm,
                          const float* __restrict__ ssl, const float* __restrict__ sdl,
                          unsigned* __restrict__ emx_m, unsigned* __restrict__ emx_l) {
  int e = blockIdx.x * blockDim.x + threadIdx.x;
  if (e >= NE + NN) return;
  int s, d; edge_sd(e, src, dst, s, d);
  atomicMax(&emx_m[d], enc_f(lrelu(ssm[s] + sdm[d])));
  atomicMax(&emx_l[d], enc_f(lrelu(ssl[s] + sdl[d])));
}

__global__ void k_gat_denom(const int* __restrict__ src, const int* __restrict__ dst,
                            const float* __restrict__ ssm, const float* __restrict__ sdm,
                            const float* __restrict__ ssl, const float* __restrict__ sdl,
                            const unsigned* __restrict__ emx_m, const unsigned* __restrict__ emx_l,
                            float* __restrict__ den_m, float* __restrict__ den_l) {
  int e = blockIdx.x * blockDim.x + threadIdx.x;
  if (e >= NE + NN) return;
  int s, d; edge_sd(e, src, dst, s, d);
  atomicAdd(&den_m[d], __expf(lrelu(ssm[s] + sdm[d]) - dec_f(emx_m[d])));
  atomicAdd(&den_l[d], __expf(lrelu(ssl[s] + sdl[d]) - dec_f(emx_l[d])));
}

__global__ void k_gat_alpha(const int* __restrict__ src, const int* __restrict__ dst,
                            const float* __restrict__ ssm, const float* __restrict__ sdm,
                            const float* __restrict__ ssl, const float* __restrict__ sdl,
                            const unsigned* __restrict__ emx_m, const unsigned* __restrict__ emx_l,
                            const float* __restrict__ den_m, const float* __restrict__ den_l,
                            float* __restrict__ alpha_m, float* __restrict__ alpha_l) {
  int e = blockIdx.x * blockDim.x + threadIdx.x;
  if (e >= NE + NN) return;
  int s, d; edge_sd(e, src, dst, s, d);
  alpha_m[e] = __expf(lrelu(ssm[s] + sdm[d]) - dec_f(emx_m[d])) / (den_m[d] + 1e-16f);
  alpha_l[e] = __expf(lrelu(ssl[s] + sdl[d]) - dec_f(emx_l[d])) / (den_l[d] + 1e-16f);
}

__global__ void k_gat_scatter(const int* __restrict__ src, const int* __restrict__ dst,
                              const float* __restrict__ Hm, const float* __restrict__ Hl,
                              const float* __restrict__ alpha_m, const float* __restrict__ alpha_l,
                              float* __restrict__ outm, float* __restrict__ outl) {
  int t = blockIdx.x * blockDim.x + threadIdx.x;
  if (t >= (NE + NN) * OUTC) return;
  int e = t >> 5, c = t & 31;
  int s, d; edge_sd(e, src, dst, s, d);
  atomicAdd(&outm[d * OUTC + c], Hm[s * OUTC + c] * alpha_m[e]);
  atomicAdd(&outl[d * OUTC + c], Hl[s * OUTC + c] * alpha_l[e]);
}

// ---------------- launch ----------------
extern "C" void kernel_launch(void* const* d_in, const int* in_sizes, int n_in,
                              void* d_out, int out_size, void* d_ws, size_t ws_size,
                              hipStream_t stream) {
  const float* x   = (const float*)d_in[0];
  const int*   ei  = (const int*)d_in[1];
  const float* ew  = (const float*)d_in[2];
  const float* Wg  = (const float*)d_in[3];
  const float* bg  = (const float*)d_in[4];
  const float* Wm  = (const float*)d_in[5];
  const float* amS = (const float*)d_in[6];
  const float* amD = (const float*)d_in[7];
  const float* bm  = (const float*)d_in[8];
  const float* Wl  = (const float*)d_in[9];
  const float* alS = (const float*)d_in[10];
  const float* alD = (const float*)d_in[11];
  const float* bl  = (const float*)d_in[12];
  const int* src = ei;
  const int* dst = ei + NE;

  // 64B-aligned bump allocator over d_ws (units: floats)
  float* wsp = (float*)d_ws;
  size_t off = 0;
  auto alloc = [&](size_t nfloats) -> float* {
    float* p = wsp + off;
    off += (nfloats + 15) & ~(size_t)15;
    return p;
  };
  float* H0   = alloc((size_t)NN * HIDC);
  float* H    = alloc((size_t)NN * HIDC);
  float* Acc  = alloc((size_t)NN * HIDC);
  float* Hm   = alloc((size_t)NN * OUTC);
  float* Hl   = alloc((size_t)NN * OUTC);
  float* dinv = alloc(NN);
  float* nrm  = alloc(NE);
  float* ssm  = alloc(NN);
  float* sdm  = alloc(NN);
  float* ssl  = alloc(NN);
  float* sdl  = alloc(NN);
  float* den_m = alloc(NN);
  float* den_l = alloc(NN);
  unsigned* emx_m = (unsigned*)alloc(NN);
  unsigned* emx_l = (unsigned*)alloc(NN);
  float* alpha_m = alloc(NE + NN);
  float* alpha_l = alloc(NE + NN);
  _Float16* Xp  = (_Float16*)alloc((size_t)NN * INC / 2);    // f16 fragments
  _Float16* Hp  = (_Float16*)alloc((size_t)NN * HIDC / 2);
  _Float16* Wgp = (_Float16*)alloc((size_t)INC * HIDC / 2);
  _Float16* Wmp = (_Float16*)alloc((size_t)HIDC * OUTC / 2);
  _Float16* Wlp = (_Float16*)alloc((size_t)HIDC * OUTC / 2);
  float* outm = (float*)d_out;
  float* outl = outm + (size_t)NN * OUTC;

  const int B = 256;

  // GCN degree norm
  k_fill_f32<<<CDIV(NN, B), B, 0, stream>>>(dinv, 1.0f, NN);       // self-loop w=1
  k_deg_scatter<<<CDIV(NE, B), B, 0, stream>>>(dinv, dst, ew, NE);
  k_dinv<<<CDIV(NN, B), B, 0, stream>>>(dinv, NN);

  // H0 = x @ W_gcn   (pack -> full-unroll WMMA, wave per 16x16 tile)
  k_pack_a<<<CDIV(NN * INC, B), B, 0, stream>>>(x, Xp, NN, INC);
  k_pack_b<<<CDIV(HIDC * INC, B), B, 0, stream>>>(Wg, Wgp, HIDC, INC);
  {
    int waves = (NN / 16) * (HIDC / 16);
    k_wmma_gemm_packed<INC><<<CDIV(waves * 32, B), B, 0, stream>>>(
        (const v16h*)Xp, (const v16h*)Wgp, H0, NN, HIDC);
  }
  k_edge_norm<<<CDIV(NE, B), B, 0, stream>>>(nrm, dinv, src, dst, ew, NE);
  k_fill_f32<<<CDIV(NN * HIDC, B), B, 0, stream>>>(Acc, 0.f, NN * HIDC);
  k_gcn_scatter<<<CDIV(NE * HIDC, B), B, 0, stream>>>(Acc, H0, nrm, src, dst, NE);
  k_gcn_finish<<<CDIV(NN * HIDC, B), B, 0, stream>>>(H, Acc, H0, dinv, bg);

  // Hm = H @ W_mean ; Hl = H @ W_logstd
  k_pack_a<<<CDIV(NN * HIDC, B), B, 0, stream>>>(H, Hp, NN, HIDC);
  k_pack_b<<<CDIV(OUTC * HIDC, B), B, 0, stream>>>(Wm, Wmp, OUTC, HIDC);
  k_pack_b<<<CDIV(OUTC * HIDC, B), B, 0, stream>>>(Wl, Wlp, OUTC, HIDC);
  {
    int waves = (NN / 16) * (OUTC / 16);
    k_wmma_gemm_packed<HIDC><<<CDIV(waves * 32, B), B, 0, stream>>>(
        (const v16h*)Hp, (const v16h*)Wmp, Hm, NN, OUTC);
    k_wmma_gemm_packed<HIDC><<<CDIV(waves * 32, B), B, 0, stream>>>(
        (const v16h*)Hp, (const v16h*)Wlp, Hl, NN, OUTC);
  }

  // attention scores, segment softmax, weighted scatter
  k_scores<<<CDIV(NN, B), B, 0, stream>>>(Hm, Hl, amS, amD, alS, alD, ssm, sdm, ssl, sdl);
  k_fill_u32<<<CDIV(2 * NN, B), B, 0, stream>>>(emx_m, 0u, 2 * NN);
  k_fill_f32<<<CDIV(2 * NN, B), B, 0, stream>>>(den_m, 0.f, 2 * NN);
  k_gat_max<<<CDIV(NE + NN, B), B, 0, stream>>>(src, dst, ssm, sdm, ssl, sdl, emx_m, emx_l);
  k_gat_denom<<<CDIV(NE + NN, B), B, 0, stream>>>(src, dst, ssm, sdm, ssl, sdl,
                                                  emx_m, emx_l, den_m, den_l);
  k_gat_alpha<<<CDIV(NE + NN, B), B, 0, stream>>>(src, dst, ssm, sdm, ssl, sdl,
                                                  emx_m, emx_l, den_m, den_l,
                                                  alpha_m, alpha_l);
  k_bias_init<<<CDIV(NN * OUTC, B), B, 0, stream>>>(outm, bm, NN, OUTC);
  k_bias_init<<<CDIV(NN * OUTC, B), B, 0, stream>>>(outl, bl, NN, OUTC);
  k_gat_scatter<<<CDIV((NE + NN) * OUTC, B), B, 0, stream>>>(src, dst, Hm, Hl,
                                                             alpha_m, alpha_l, outm, outl);
}
